// GNN_Network_infer_3324304687116
// MI455X (gfx1250) — compile-verified, matrix operand order checked by reading, not run on hardware
//
#include <hip/hip_runtime.h>
#include <hip/hip_bf16.h>

typedef __attribute__((ext_vector_type(16))) __bf16 v16bf;
typedef __attribute__((ext_vector_type(8)))  float  v8f;

#define NN 10000
#define EE 160000
#define DD 32

// Native v_tanh_f32 on gfx1250 if available, else branch-free exp2/rcp form.
__device__ __forceinline__ float fast_tanh(float x) {
#if __has_builtin(__builtin_amdgcn_tanhf)
    return __builtin_amdgcn_tanhf(x);
#else
    float z = __builtin_amdgcn_exp2f(x * 2.885390081777927f); // 2*log2(e)
    return 1.0f - 2.0f * __builtin_amdgcn_rcpf(z + 1.0f);
#endif
}

// ---------------------------------------------------------------- utilities
__global__ __launch_bounds__(256) void zero_kernel(float* p, int n) {
    int i = blockIdx.x * 256 + threadIdx.x;
    if (i < n) p[i] = 0.0f;
}

__global__ __launch_bounds__(256) void deg_kernel(const int* __restrict__ dst, float* deg) {
    int e = blockIdx.x * 256 + threadIdx.x;
    if (e < EE) atomicAdd(&deg[dst[e]], 1.0f);
}

__global__ __launch_bounds__(256) void invdeg_kernel(const float* __restrict__ deg, float* inv) {
    int i = blockIdx.x * 256 + threadIdx.x;
    if (i < NN) inv[i] = 1.0f / fmaxf(deg[i], 1.0f);
}

// Pre-swizzle Wm into the WMMA B-operand layout (bf16) and bake the MLP bias
// into "bias-pair" dwords that drop straight into B's K=16 padding slot:
//   weights: wpW[(blkbase+blk)*256 + lane*8 + r] = bf16(Wm[row][col])
//   bias   : wpB[(blkbase+blk)*32 + lane] = lane<16 ? {hi:0, lo:bf16(bm[col])} : 0
__global__ __launch_bounds__(256)
void wprep_kernel(const float* __restrict__ Wm0, const float* __restrict__ Wm1,
                  const float* __restrict__ Wm2, const float* __restrict__ Wm3,
                  const float* __restrict__ bm0, const float* __restrict__ bm1,
                  const float* __restrict__ bm2, const float* __restrict__ bm3,
                  unsigned short* __restrict__ wpW, unsigned int* __restrict__ wpB) {
    int idx = blockIdx.x * 256 + threadIdx.x;
    if (idx < 49664) {                             // weights: 3*64*256 + 2*256
        int l, rem;
        if (idx < 49152) { l = idx / 16384; rem = idx % 16384; }
        else             { l = 3;           rem = idx - 49152; }
        int blk  = rem >> 8;
        int lane = (rem >> 3) & 31;
        int r    = rem & 7;
        int row  = ((lane & 16) ? 8 : 0) + r;
        const float* W; int ncol, col;
        if (l < 3) { W = (l == 0 ? Wm0 : (l == 1 ? Wm1 : Wm2)); ncol = 1024;
                     col = (blk >> 1) * 32 + (blk & 1) * 16 + (lane & 15); }
        else       { W = Wm3; ncol = 32; col = blk * 16 + (lane & 15); }
        __bf16 bv = (__bf16)W[row * ncol + col];
        unsigned short bits;
        __builtin_memcpy(&bits, &bv, 2);
        wpW[idx] = bits;
    } else if (idx < 49664 + 6208) {               // bias dwords: (3*64+2)*32
        int bi = idx - 49664;
        int gblk = bi >> 5;                        // 0..193
        int lane = bi & 31;
        int l, blk;
        if (gblk < 192) { l = gblk / 64; blk = gblk % 64; }
        else            { l = 3;         blk = gblk - 192; }
        unsigned int dw = 0;
        if (lane < 16) {
            const float* B; int col;
            if (l < 3) { B = (l == 0 ? bm0 : (l == 1 ? bm1 : bm2));
                         col = (blk >> 1) * 32 + (blk & 1) * 16 + lane; }
            else       { B = bm3; col = blk * 16 + lane; }
            __bf16 bv = (__bf16)B[col];
            unsigned short bits;
            __builtin_memcpy(&bits, &bv, 2);
            dw = (unsigned int)bits;               // low bf16 = bias, high = 0
        }
        wpB[bi] = dw;
    }
}

// ------------------------------------------- fused edge kernel, oc = 32
__global__ __launch_bounds__(256)
void edge_conv32(const float* __restrict__ x, const int* __restrict__ src,
                 const int* __restrict__ dst, const float* __restrict__ ea,
                 const unsigned short* __restrict__ wprep,
                 const unsigned int* __restrict__ wbias,
                 float* __restrict__ agg) {
    __shared__ unsigned short lws[64 * 32 * 8];   // 32 KB: Wm bf16 B-layout
    __shared__ unsigned int   lbp[64 * 32];       // 8 KB : bias-pair dwords
    __shared__ float lxs[8 * 16 * 32];            // 16 KB: gathered x[src]
    int tid = threadIdx.x;
    {
        const uint4* wsrc = (const uint4*)wprep;
        uint4* wdst = (uint4*)lws;
        #pragma unroll
        for (int i = 0; i < 8; ++i) wdst[tid + i * 256] = wsrc[tid + i * 256];
        #pragma unroll
        for (int i = 0; i < 8; ++i) lbp[tid + i * 256] = wbias[tid + i * 256];
    }
    int wave = tid >> 5, lane = tid & 31;
    int laneLo = lane & 15, laneHi = lane >> 4;
    int e0  = (blockIdx.x * 8 + wave) * 16;
    int myE = e0 + laneLo;
    int s = src[myE], d = dst[myE];

    // A operand: K 0..15 = edge_attr, K=16 = 1.0 (bias lane), rest zero-pad.
    v16bf a;
    {
        const float4* eap = (const float4*)(ea + (size_t)myE * 16 + laneHi * 8);
        float4 f0 = eap[0], f1 = eap[1];
        a[0]=(__bf16)f0.x; a[1]=(__bf16)f0.y; a[2]=(__bf16)f0.z; a[3]=(__bf16)f0.w;
        a[4]=(__bf16)f1.x; a[5]=(__bf16)f1.y; a[6]=(__bf16)f1.z; a[7]=(__bf16)f1.w;
        a[8] = laneHi ? (__bf16)0.0f : (__bf16)1.0f;   // K=16 "ones" column
        #pragma unroll
        for (int k = 9; k < 16; ++k) a[k] = (__bf16)0.0f;
    }
    // gather x[src[e]] rows into LDS (per-wave region)
    float* xsw = lxs + wave * 16 * 32;
    #pragma unroll
    for (int k = 0; k < 16; ++k) {
        int sk = __shfl(s, k, 32);
        xsw[k * 32 + lane] = x[(size_t)sk * 32 + lane];
    }
    __syncthreads();

    v8f m0 = {0.f,0.f,0.f,0.f,0.f,0.f,0.f,0.f};
    v8f m1 = {0.f,0.f,0.f,0.f,0.f,0.f,0.f,0.f};
    const v8f cz = {0.f,0.f,0.f,0.f,0.f,0.f,0.f,0.f};

    for (int i = 0; i < 32; ++i) {
        float xsv[8];
        #pragma unroll
        for (int v = 0; v < 8; ++v) xsv[v] = xsw[(v + laneHi * 8) * 32 + i];
        #pragma unroll
        for (int h = 0; h < 2; ++h) {
            int blk = i * 2 + h;
            uint4 wbits = ((const uint4*)lws)[blk * 32 + lane];
            unsigned int bp = lbp[blk * 32 + lane];    // K=16 slot: {0, bias}
            union { uint4 u; __bf16 b8[8]; } cv; cv.u = wbits;
            union { unsigned int u; __bf16 h2[2]; } c2; c2.u = bp;
            v16bf b;
            #pragma unroll
            for (int k = 0; k < 8; ++k) b[k] = cv.b8[k];
            b[8] = c2.h2[0]; b[9] = c2.h2[1];
            #pragma unroll
            for (int k = 10; k < 16; ++k) b[k] = (__bf16)0.0f;
            // D = A*B (+ inline-0 C): theta + bias via the K=16 ones column.
            v8f t = __builtin_amdgcn_wmma_f32_16x16x32_bf16(
                false, a, false, b, (short)0, cz, false, false);
            #pragma unroll
            for (int v = 0; v < 8; ++v) {
                float th = fast_tanh(t[v]);
                if (h == 0) m0[v] += xsv[v] * th; else m1[v] += xsv[v] * th;
            }
        }
    }
    // scatter-add: lane holds (edge v+laneHi*8, col laneLo / laneLo+16)
    #pragma unroll
    for (int v = 0; v < 8; ++v) {
        int dv = __shfl(d, v + laneHi * 8, 32);
        atomicAdd(&agg[(size_t)dv * 32 + laneLo],      m0[v]);
        atomicAdd(&agg[(size_t)dv * 32 + 16 + laneLo], m1[v]);
    }
}

// ------------------------------------------- fused edge kernel, oc = 1 (readout)
__global__ __launch_bounds__(256)
void edge_conv1(const float* __restrict__ x, const int* __restrict__ src,
                const int* __restrict__ dst, const float* __restrict__ ea,
                const unsigned short* __restrict__ wprep,
                const unsigned int* __restrict__ wbias,
                float* __restrict__ agg1) {
    __shared__ unsigned short lws[2 * 32 * 8];    // 1 KB
    __shared__ unsigned int   lbp[2 * 32];        // 256 B
    __shared__ float lxs[8 * 16 * 32];            // 16 KB
    __shared__ float lred[8 * 8 * 32];            // 8 KB
    int tid = threadIdx.x;
    if (tid < 64) ((uint4*)lws)[tid] = ((const uint4*)wprep)[tid];
    if (tid < 64) lbp[tid] = wbias[tid];
    int wave = tid >> 5, lane = tid & 31;
    int laneLo = lane & 15, laneHi = lane >> 4;
    int e0  = (blockIdx.x * 8 + wave) * 16;
    int myE = e0 + laneLo;
    int s = src[myE];
    v16bf a;
    {
        const float4* eap = (const float4*)(ea + (size_t)myE * 16 + laneHi * 8);
        float4 f0 = eap[0], f1 = eap[1];
        a[0]=(__bf16)f0.x; a[1]=(__bf16)f0.y; a[2]=(__bf16)f0.z; a[3]=(__bf16)f0.w;
        a[4]=(__bf16)f1.x; a[5]=(__bf16)f1.y; a[6]=(__bf16)f1.z; a[7]=(__bf16)f1.w;
        a[8] = laneHi ? (__bf16)0.0f : (__bf16)1.0f;
        #pragma unroll
        for (int k = 9; k < 16; ++k) a[k] = (__bf16)0.0f;
    }
    float* xsw = lxs + wave * 16 * 32;
    #pragma unroll
    for (int k = 0; k < 16; ++k) {
        int sk = __shfl(s, k, 32);
        xsw[k * 32 + lane] = x[(size_t)sk * 32 + lane];
    }
    __syncthreads();

    const v8f cz = {0.f,0.f,0.f,0.f,0.f,0.f,0.f,0.f};
    float p[8] = {0.f,0.f,0.f,0.f,0.f,0.f,0.f,0.f};
    #pragma unroll
    for (int h = 0; h < 2; ++h) {
        uint4 wbits = ((const uint4*)lws)[h * 32 + lane];
        unsigned int bp = lbp[h * 32 + lane];
        union { uint4 u; __bf16 b8[8]; } cv; cv.u = wbits;
        union { unsigned int u; __bf16 h2[2]; } c2; c2.u = bp;
        v16bf b;
        #pragma unroll
        for (int k = 0; k < 8; ++k) b[k] = cv.b8[k];
        b[8] = c2.h2[0]; b[9] = c2.h2[1];
        #pragma unroll
        for (int k = 10; k < 16; ++k) b[k] = (__bf16)0.0f;
        int ich = h * 16 + laneLo;                 // input-channel column
        v8f t = __builtin_amdgcn_wmma_f32_16x16x32_bf16(
            false, a, false, b, (short)0, cz, false, false);
        #pragma unroll
        for (int v = 0; v < 8; ++v)
            p[v] += fast_tanh(t[v]) * xsw[(v + laneHi * 8) * 32 + ich];
    }
    float* rw = lred + wave * 8 * 32;
    #pragma unroll
    for (int v = 0; v < 8; ++v) rw[v * 32 + lane] = p[v];
    __syncthreads();
    if (lane < 16) {
        int ev = lane & 7, half = lane >> 3;
        float sum = 0.f;
        #pragma unroll
        for (int l = 0; l < 16; ++l) sum += rw[ev * 32 + half * 16 + l];
        atomicAdd(&agg1[dst[e0 + lane]], sum);
    }
}

// ---------------- node: h = agg*invdeg + x@Wr + br, + BN batch stats
__global__ __launch_bounds__(256)
void node_a(const float* __restrict__ x, const float* __restrict__ agg,
            const float* __restrict__ invdeg, const float* __restrict__ Wr,
            const float* __restrict__ br, float* __restrict__ h,
            float* __restrict__ stats) {
    __shared__ float lW[1024];
    __shared__ float lx[256];
    __shared__ float lh[256];
    int tid = threadIdx.x;
    #pragma unroll
    for (int i = 0; i < 4; ++i) lW[tid + i * 256] = Wr[tid + i * 256];
    int c = tid & 31, r = tid >> 5;
    int n = blockIdx.x * 8 + r;
    lx[tid] = x[(size_t)blockIdx.x * 256 + tid];  // coalesced: 8 node rows
    __syncthreads();
    float acc = 0.f;
    #pragma unroll
    for (int i = 0; i < 32; ++i) acc += lx[r * 32 + i] * lW[i * 32 + c];
    float hv = agg[(size_t)n * 32 + c] * invdeg[n] + acc + br[c];
    h[(size_t)n * 32 + c] = hv;
    lh[tid] = hv;
    __syncthreads();
    if (r == 0) {
        float s = 0.f, s2 = 0.f;
        #pragma unroll
        for (int j = 0; j < 8; ++j) { float v = lh[j * 32 + c]; s += v; s2 += v * v; }
        atomicAdd(&stats[c], s);
        atomicAdd(&stats[32 + c], s2);
    }
}

// ---------------- BN normalize + ReLU + residual
__global__ __launch_bounds__(256)
void node_b(const float* __restrict__ x, const float* __restrict__ h,
            const float* __restrict__ stats, const float* __restrict__ g,
            const float* __restrict__ be, float* __restrict__ xout) {
    int idx = blockIdx.x * 256 + threadIdx.x;     // < N*32 exactly
    int c = idx & 31;
    float mu  = stats[c] * (1.0f / NN);
    float var = stats[32 + c] * (1.0f / NN) - mu * mu;
    float rs  = rsqrtf(var + 1e-5f);
    float v = (h[idx] - mu) * rs * g[c] + be[c];
    xout[idx] = x[idx] + fmaxf(v, 0.0f);
}

// ---------------- final readout node side
__global__ __launch_bounds__(256)
void final_node(const float* __restrict__ x, const float* __restrict__ agg1,
                const float* __restrict__ invdeg, const float* __restrict__ Wr3,
                const float* __restrict__ br3, float* __restrict__ out) {
    __shared__ float lw[32];
    if (threadIdx.x < 32) lw[threadIdx.x] = Wr3[threadIdx.x];
    __syncthreads();
    int n = blockIdx.x * 256 + threadIdx.x;
    if (n >= NN) return;
    float acc = 0.f;
    #pragma unroll
    for (int i = 0; i < 32; ++i) acc += x[(size_t)n * 32 + i] * lw[i];
    out[n] = agg1[n] * invdeg[n] + acc + br3[0];
}

// ============================================================ launcher
extern "C" void kernel_launch(void* const* d_in, const int* in_sizes, int n_in,
                              void* d_out, int out_size, void* d_ws, size_t ws_size,
                              hipStream_t stream) {
    const float* x0 = (const float*)d_in[0];
    const int*   ei = (const int*)d_in[1];
    const float* ea = (const float*)d_in[2];
    const float *Wm[4], *bm[4], *Wr[4], *br[4];
    int p = 3;
    for (int l = 0; l < 4; ++l) {
        Wm[l] = (const float*)d_in[p++]; bm[l] = (const float*)d_in[p++];
        Wr[l] = (const float*)d_in[p++]; br[l] = (const float*)d_in[p++];
    }
    const float *g[3], *be[3];
    for (int l = 0; l < 3; ++l) { g[l] = (const float*)d_in[p++]; be[l] = (const float*)d_in[p++]; }
    float* out = (float*)d_out;
    const int* srcp = ei;
    const int* dstp = ei + EE;

    // workspace carve (all offsets 16B-aligned)
    char* w = (char*)d_ws;
    float* xA     = (float*)w;                w += (size_t)NN * 32 * 4;
    float* xB     = (float*)w;                w += (size_t)NN * 32 * 4;
    float* hbuf   = (float*)w;                w += (size_t)NN * 32 * 4;
    float* agg    = (float*)w;                w += (size_t)NN * 32 * 4;
    float* deg    = (float*)w;                w += (size_t)NN * 4;
    float* invd   = (float*)w;                w += (size_t)NN * 4;
    float* agg1   = (float*)w;                w += (size_t)NN * 4;
    float* stats  = (float*)w;                w += 256;
    unsigned short* wpW = (unsigned short*)w; w += 49664 * 2;   // bf16 weights
    unsigned int*   wpB = (unsigned int*)w;   w += 6208 * 4;    // bias dwords

    // degree (dst-only, layer invariant)
    zero_kernel<<<(NN + 255) / 256, 256, 0, stream>>>(deg, NN);
    deg_kernel<<<EE / 256, 256, 0, stream>>>(dstp, deg);
    invdeg_kernel<<<(NN + 255) / 256, 256, 0, stream>>>(deg, invd);
    // Wm/bm -> bf16 WMMA-B layout + bias-pair dwords
    wprep_kernel<<<(49664 + 6208 + 255) / 256, 256, 0, stream>>>(
        Wm[0], Wm[1], Wm[2], Wm[3], bm[0], bm[1], bm[2], bm[3], wpW, wpB);

    const float* xin = x0;
    float* xout = xA;
    for (int l = 0; l < 3; ++l) {
        zero_kernel<<<(NN * 32 + 255) / 256, 256, 0, stream>>>(agg, NN * 32);
        zero_kernel<<<1, 256, 0, stream>>>(stats, 64);
        edge_conv32<<<EE / 128, 256, 0, stream>>>(xin, srcp, dstp, ea,
                                                  wpW + 16384 * l, wpB + 2048 * l, agg);
        node_a<<<NN / 8, 256, 0, stream>>>(xin, agg, invd, Wr[l], br[l], hbuf, stats);
        node_b<<<(NN * 32) / 256, 256, 0, stream>>>(xin, hbuf, stats, g[l], be[l], xout);
        xin = xout;
        xout = (xout == xA) ? xB : xA;
    }
    zero_kernel<<<(NN + 255) / 256, 256, 0, stream>>>(agg1, NN);
    edge_conv1<<<EE / 128, 256, 0, stream>>>(xin, srcp, dstp, ea,
                                             wpW + 49152, wpB + 6144, agg1);
    final_node<<<(NN + 255) / 256, 256, 0, stream>>>(xin, agg1, invd, Wr[3], br[3], out);
}
